// Block_39316130628178
// MI455X (gfx1250) — matrix operand order repaired and checked by reading, not compiled, not run on hardware
//
#include <hip/hip_runtime.h>
#include <hip/hip_bf16.h>

// ---------------------------------------------------------------------------
// Transformer decoder block for MI455X (gfx1250, wave32, WMMA).
// bf16 WMMA (f32 accum) for all GEMMs + flash attention; f32 softmax/LN/GELU.
// Async global->LDS staging (ASYNCcnt) with double buffering in the GEMM.
// ---------------------------------------------------------------------------

typedef __attribute__((ext_vector_type(16))) __bf16 v16bf;
typedef __attribute__((ext_vector_type(8)))  __bf16 bf16x8;
typedef __attribute__((ext_vector_type(8)))  float  v8f;
typedef int v4i_ __attribute__((vector_size(16)));   // matches builtin param type

#define B_      2
#define S_      2048
#define D_      1024
#define H_      16
#define DH_     64
#define INNER_  1024
#define FF_     4096
#define M_      (B_*S_)       // 4096 rows of activations

__device__ __forceinline__ v8f vzero8() {
  return (v8f){0.f,0.f,0.f,0.f,0.f,0.f,0.f,0.f};
}

__device__ __forceinline__ v8f wmma_bf16(v16bf a, v16bf b, v8f c) {
  // D = A(16x32 bf16) * B(32x16 bf16) + C(16x16 f32)
  return __builtin_amdgcn_wmma_f32_16x16x32_bf16(false, a, false, b, (short)0, c,
                                                 false, false);
}

// Combine two aligned 8-half LDS chunks into one 16-half fragment register set.
__device__ __forceinline__ v16bf ldfrag(const __bf16* p0, const __bf16* p1) {
  bf16x8 lo = *(const bf16x8*)p0;
  bf16x8 hi = *(const bf16x8*)p1;
  return __builtin_shufflevector(lo, hi, 0,1,2,3,4,5,6,7,8,9,10,11,12,13,14,15);
}

// ---------------------------------------------------------------------------
// CDNA5 async global->LDS (16B per lane), tracked by ASYNCcnt.
// Builtin when the toolchain has it, else raw ISA via inline asm.
// ---------------------------------------------------------------------------
#if defined(__has_builtin)
#if __has_builtin(__builtin_amdgcn_global_load_async_to_lds_b128)
#define ASYNC_LDS_BUILTIN 1
#endif
#if __has_builtin(__builtin_amdgcn_s_wait_asynccnt)
#define ASYNC_WAIT_BUILTIN 1
#endif
#endif

__device__ __forceinline__ void async_copy16(const __bf16* g, __bf16* l) {
#ifdef ASYNC_LDS_BUILTIN
  __builtin_amdgcn_global_load_async_to_lds_b128(
      (__attribute__((address_space(1))) v4i_*)g,
      (__attribute__((address_space(3))) v4i_*)l, 0, 0);
#else
  // VDST = LDS byte address (low 32 bits of flat shared addr), VADDR = 64b GVA.
  asm volatile("global_load_async_to_lds_b128 %0, %1, off"
               :: "v"((unsigned)(__UINTPTR_TYPE__)l), "v"(g)
               : "memory");
#endif
}

__device__ __forceinline__ void wait_async0() {
#ifdef ASYNC_WAIT_BUILTIN
  __builtin_amdgcn_s_wait_asynccnt(0);
#else
  asm volatile("s_wait_asynccnt 0x0" ::: "memory");
#endif
}

// ---------------------------------------------------------------------------
// f32 -> bf16 copy
// ---------------------------------------------------------------------------
__global__ void to_bf16(const float* __restrict__ in, __bf16* __restrict__ out,
                        size_t n) {
  size_t i = (size_t)blockIdx.x * blockDim.x + threadIdx.x;
  size_t stride = (size_t)gridDim.x * blockDim.x;
  for (; i < n; i += stride) out[i] = (__bf16)in[i];
}

// f32 (K x N) -> bf16 (N x K): weights pre-transposed so GEMM B-tiles are
// coalesced in global and land in LDS in WMMA B-fragment (n-major, contig-K).
__global__ __launch_bounds__(256)
void transpose_to_bf16(const float* __restrict__ in, __bf16* __restrict__ out,
                       int K, int N) {
  __shared__ float t[32][33];
  const int tx = threadIdx.x & 31, ty = threadIdx.x >> 5;   // 32 x 8
  const int k0 = blockIdx.y * 32, n0 = blockIdx.x * 32;
#pragma unroll
  for (int i = 0; i < 4; i++)
    t[ty + i*8][tx] = in[(size_t)(k0 + ty + i*8) * N + n0 + tx];
  __syncthreads();
#pragma unroll
  for (int i = 0; i < 4; i++)
    out[(size_t)(n0 + ty + i*8) * K + k0 + tx] = (__bf16)t[tx][ty + i*8];
}

// ---------------------------------------------------------------------------
// WMMA GEMM:  C(MxN) = A(MxK, bf16 row-major) * Bt(NxK, bf16 row-major)^T
// Block tile 128x128, K-step 32 (= WMMA K).  8 waves (2x4), wave tile 64x32.
// Double-buffered LDS fed by async global->LDS; two barriers per K-step.
// EPI: 0 = none, 1 = +bias, 2 = +bias + exact GELU.
// ---------------------------------------------------------------------------
template <int EPI, typename OutT>
__global__ __launch_bounds__(256)
void gemm_wmma(const __bf16* __restrict__ A, const __bf16* __restrict__ Bt,
               const float* __restrict__ bias, OutT* __restrict__ C,
               int M, int N, int K) {
  constexpr int BM = 128, BN = 128, BK = 32;
  constexpr int LDT = BK + 8;                 // 40 halves = 80B rows (16B aligned)
  __shared__ __bf16 As[2][BM * LDT];
  __shared__ __bf16 Bs[2][BN * LDT];

  const int tid  = threadIdx.x;
  const int lane = tid & 31, wave = tid >> 5;
  const int g = lane >> 4, ln = lane & 15;    // lane-group, lane-in-group
  const int wrow = wave >> 2, wcol = wave & 3;
  const int m0 = blockIdx.y * BM, n0 = blockIdx.x * BN;

  // Per-thread staging coordinates: 512 16B-chunks per tile, 2 per thread.
  const int r0 = tid >> 2,          o0 = (tid & 3) * 8;
  const int r1 = (tid + 256) >> 2,  o1 = ((tid + 256) & 3) * 8;
  const __bf16* gA0 = &A [(size_t)(m0 + r0) * K + o0];
  const __bf16* gA1 = &A [(size_t)(m0 + r1) * K + o1];
  const __bf16* gB0 = &Bt[(size_t)(n0 + r0) * K + o0];
  const __bf16* gB1 = &Bt[(size_t)(n0 + r1) * K + o1];

  auto issue = [&](int buf, int k0) {
    async_copy16(gA0 + k0, &As[buf][r0 * LDT + o0]);
    async_copy16(gA1 + k0, &As[buf][r1 * LDT + o1]);
    async_copy16(gB0 + k0, &Bs[buf][r0 * LDT + o0]);
    async_copy16(gB1 + k0, &Bs[buf][r1 * LDT + o1]);
  };

  v8f acc[4][2];
#pragma unroll
  for (int i = 0; i < 4; i++)
#pragma unroll
    for (int j = 0; j < 2; j++) acc[i][j] = vzero8();

  const int T = K / BK;
  issue(0, 0);

  for (int it = 0; it < T; it++) {
    const int cur = it & 1;
    wait_async0();        // my tile-it loads have landed in LDS
    __syncthreads();      // everyone's have; prev compute done -> cur^1 free
    if (it + 1 < T) issue(cur ^ 1, (it + 1) * BK);

    // A fragment: halves 0..15 contiguous at K = g*16.
    v16bf af[4];
#pragma unroll
    for (int rt = 0; rt < 4; rt++) {
      const __bf16* rp = &As[cur][(wrow * 64 + rt * 16 + ln) * LDT];
      af[rt] = ldfrag(rp + g * 16, rp + g * 16 + 8);
    }
    // B fragment: per-lane column, halves 0..15 contiguous at K = g*16.
    v16bf bfr[2];
#pragma unroll
    for (int ct = 0; ct < 2; ct++) {
      const __bf16* cp = &Bs[cur][(wcol * 32 + ct * 16 + ln) * LDT];
      bfr[ct] = ldfrag(cp + g * 16, cp + g * 16 + 8);
    }
#pragma unroll
    for (int rt = 0; rt < 4; rt++)
#pragma unroll
      for (int ct = 0; ct < 2; ct++)
        acc[rt][ct] = wmma_bf16(af[rt], bfr[ct], acc[rt][ct]);

    __syncthreads();      // compute done before next iteration's overwrite gate
  }

  // Epilogue.  C/D layout: row = g*8 + r, col = ln.
#pragma unroll
  for (int rt = 0; rt < 4; rt++) {
#pragma unroll
    for (int ct = 0; ct < 2; ct++) {
      const int n = n0 + wcol * 32 + ct * 16 + ln;
      float bv = 0.f;
      if constexpr (EPI >= 1) bv = bias[n];
#pragma unroll
      for (int r = 0; r < 8; r++) {
        const int m = m0 + wrow * 64 + rt * 16 + g * 8 + r;
        float v = acc[rt][ct][r] + bv;
        if constexpr (EPI == 2)
          v = 0.5f * v * (1.0f + erff(v * 0.70710678118654752f));
        C[(size_t)m * N + n] = (OutT)v;
      }
    }
  }
}

// ---------------------------------------------------------------------------
// Flash attention, one (b, h, 64-query tile) per block.  4 waves x 16 q-rows.
// K/V consumed in 32-key blocks; K staged via async global->LDS.
// Key-mask per reference: sim = mask ? -big.
// ---------------------------------------------------------------------------
__global__ __launch_bounds__(128)
void attn_wmma(const __bf16* __restrict__ Q, const __bf16* __restrict__ Km,
               const __bf16* __restrict__ Vm,
               const unsigned char* __restrict__ mask,
               __bf16* __restrict__ O) {
  constexpr int QT = 64, JB = 32;
  constexpr int LQ = DH_ + 8;   // 72 halves (144B, 16B aligned)
  constexpr int LV = JB + 8;    // 40 halves (80B, 16B aligned)
  constexpr float SCALE = 0.125f;   // 64^-0.5
  constexpr float NEG   = -3.0e38f;

  __shared__ __bf16 Qs[QT * LQ];       // Q tile, row-major (i x d)
  __shared__ __bf16 Ks[JB * LQ];       // K block, row-major (j x d)
  __shared__ __bf16 Vst[DH_ * LV];     // V block transposed (d x j)
  __shared__ __bf16 Ps[4 * 16 * LV];   // per-wave P tiles (16 x 32)
  __shared__ float  maskb[JB];

  const int tid  = threadIdx.x;
  const int lane = tid & 31, wave = tid >> 5;
  const int g = lane >> 4, ln = lane & 15;
  const int qt = blockIdx.x, h = blockIdx.y, b = blockIdx.z;
  const size_t baseQ  = ((size_t)b * S_ + (size_t)qt * QT) * INNER_ + h * DH_;
  const size_t baseKV = (size_t)b * S_ * INNER_ + h * DH_;

  // Stage Q tile (64x64 halves = 512 16B-chunks, 4 per thread) via async.
#pragma unroll
  for (int i = 0; i < 4; i++) {
    int c = tid + i * 128;
    int row = c >> 3, off = (c & 7) * 8;
    async_copy16(&Q[baseQ + (size_t)row * INNER_ + off], &Qs[row * LQ + off]);
  }
  wait_async0();
  __syncthreads();

  // Persistent Q A-fragments for this wave's 16 rows (d 0..31 and d 32..63).
  const __bf16* qrp = &Qs[(wave * 16 + ln) * LQ];
  const v16bf qa0 = ldfrag(qrp + g * 16,      qrp + g * 16 + 8);
  const v16bf qa1 = ldfrag(qrp + 32 + g * 16, qrp + 32 + g * 16 + 8);

  float mrow[8], lrow[8];
  v8f oacc[4];
#pragma unroll
  for (int r = 0; r < 8; r++) { mrow[r] = -INFINITY; lrow[r] = 0.f; }
#pragma unroll
  for (int ct = 0; ct < 4; ct++) oacc[ct] = vzero8();

  for (int jb = 0; jb < S_; jb += JB) {
    __syncthreads();   // previous iteration's LDS reads complete
    // Stage K block (j x d) via async: layout == K^T B-fragment layout.
#pragma unroll
    for (int i = 0; i < 2; i++) {
      int c = tid + i * 128;
      int row = c >> 3, off = (c & 7) * 8;
      async_copy16(&Km[baseKV + (size_t)(jb + row) * INNER_ + off],
                   &Ks[row * LQ + off]);
    }
    // Stage V block transposed (d x j) for the PV B-fragments (VGPR transit).
#pragma unroll
    for (int i = 0; i < 2; i++) {
      int c = tid + i * 128;
      int j = c >> 3, dc = (c & 7) * 8;
      bf16x8 vv = *(const bf16x8*)&Vm[baseKV + (size_t)(jb + j) * INNER_ + dc];
#pragma unroll
      for (int e = 0; e < 8; e++) Vst[(dc + e) * LV + j] = vv[e];
    }
    if (tid < JB) maskb[tid] = mask[(size_t)b * S_ + jb + tid] ? 1.f : 0.f;
    wait_async0();
    __syncthreads();

    // sim = Q * K^T for two 16x16 column tiles (j 0..15, 16..31).
    v8f s0 = vzero8(), s1 = vzero8();
    {
      const __bf16* kp = &Ks[ln * LQ];
      s0 = wmma_bf16(qa0, ldfrag(kp + g * 16,      kp + g * 16 + 8),      s0);
      s0 = wmma_bf16(qa1, ldfrag(kp + 32 + g * 16, kp + 32 + g * 16 + 8), s0);
    }
    {
      const __bf16* kp = &Ks[(16 + ln) * LQ];
      s1 = wmma_bf16(qa0, ldfrag(kp + g * 16,      kp + g * 16 + 8),      s1);
      s1 = wmma_bf16(qa1, ldfrag(kp + 32 + g * 16, kp + 32 + g * 16 + 8), s1);
    }

    const bool mk0 = maskb[ln] != 0.f;
    const bool mk1 = maskb[16 + ln] != 0.f;
    float sv0[8], sv1[8], rmax[8];
#pragma unroll
    for (int r = 0; r < 8; r++) {
      sv0[r] = mk0 ? NEG : s0[r] * SCALE;
      sv1[r] = mk1 ? NEG : s1[r] * SCALE;
      rmax[r] = fmaxf(sv0[r], sv1[r]);
    }
    // Row reductions stay inside each 16-lane group (rows g*8 .. g*8+7).
#pragma unroll
    for (int off = 8; off >= 1; off >>= 1)
#pragma unroll
      for (int r = 0; r < 8; r++)
        rmax[r] = fmaxf(rmax[r], __shfl_xor(rmax[r], off, 32));

    float alpha[8], p0[8], p1[8], psum[8];
#pragma unroll
    for (int r = 0; r < 8; r++) {
      float mnew = fmaxf(mrow[r], rmax[r]);
      alpha[r] = __expf(mrow[r] - mnew);
      p0[r]    = __expf(sv0[r] - mnew);
      p1[r]    = __expf(sv1[r] - mnew);
      psum[r]  = p0[r] + p1[r];
      mrow[r]  = mnew;
    }
#pragma unroll
    for (int off = 8; off >= 1; off >>= 1)
#pragma unroll
      for (int r = 0; r < 8; r++)
        psum[r] += __shfl_xor(psum[r], off, 32);
#pragma unroll
    for (int r = 0; r < 8; r++) lrow[r] = lrow[r] * alpha[r] + psum[r];
#pragma unroll
    for (int ct = 0; ct < 4; ct++)
#pragma unroll
      for (int r = 0; r < 8; r++) oacc[ct][r] = oacc[ct][r] * alpha[r];

    // Re-layout P (C-layout -> A-fragment) through LDS.
    __bf16* pw = &Ps[wave * 16 * LV];
#pragma unroll
    for (int r = 0; r < 8; r++) {
      pw[(g * 8 + r) * LV + ln]      = (__bf16)p0[r];
      pw[(g * 8 + r) * LV + 16 + ln] = (__bf16)p1[r];
    }
    __syncthreads();

    const __bf16* pr = &Ps[(wave * 16 + ln) * LV];
    const v16bf pa = ldfrag(pr + g * 16, pr + g * 16 + 8);
#pragma unroll
    for (int ct = 0; ct < 4; ct++) {
      const __bf16* vp = &Vst[(ct * 16 + ln) * LV];
      oacc[ct] = wmma_bf16(pa, ldfrag(vp + g * 16, vp + g * 16 + 8), oacc[ct]);
    }
  }

  // Normalize and write (b, i, h, d) as bf16 (feeds the output projection).
#pragma unroll
  for (int ct = 0; ct < 4; ct++)
#pragma unroll
    for (int r = 0; r < 8; r++) {
      const int row = qt * QT + wave * 16 + g * 8 + r;
      O[((size_t)b * S_ + row) * INNER_ + h * DH_ + ct * 16 + ln] =
          (__bf16)(oacc[ct][r] / lrow[r]);
    }
}

// ---------------------------------------------------------------------------
// y = LayerNorm(x + r) * gamma + beta  (biased variance, eps 1e-5).
// Writes f32 (residual chain) and optional bf16 (next GEMM A operand).
// ---------------------------------------------------------------------------
__global__ __launch_bounds__(256)
void add_ln(const float* __restrict__ X, const float* __restrict__ R,
            const float* __restrict__ gamma, const float* __restrict__ beta,
            float* __restrict__ Yf, __bf16* __restrict__ Ybf) {
  __shared__ float r1[256], r2[256];
  const int row = blockIdx.x, tid = threadIdx.x;
  const float* x  = X + (size_t)row * D_;
  const float* rr = R + (size_t)row * D_;
  float v[4], s = 0.f, ss = 0.f;
#pragma unroll
  for (int i = 0; i < 4; i++) {
    v[i] = x[tid + i * 256] + rr[tid + i * 256];
    s += v[i];
    ss += v[i] * v[i];
  }
  r1[tid] = s; r2[tid] = ss;
  __syncthreads();
  for (int st = 128; st > 0; st >>= 1) {
    if (tid < st) { r1[tid] += r1[tid + st]; r2[tid] += r2[tid + st]; }
    __syncthreads();
  }
  const float mean = r1[0] * (1.f / D_);
  const float var  = r2[0] * (1.f / D_) - mean * mean;
  const float rstd = rsqrtf(var + 1e-5f);
#pragma unroll
  for (int i = 0; i < 4; i++) {
    const int c = tid + i * 256;
    const float y = (v[i] - mean) * rstd * gamma[c] + beta[c];
    Yf[(size_t)row * D_ + c] = y;
    if (Ybf) Ybf[(size_t)row * D_ + c] = (__bf16)y;
  }
}

// ---------------------------------------------------------------------------
// Host orchestration
// ---------------------------------------------------------------------------
extern "C" void kernel_launch(void* const* d_in, const int* in_sizes, int n_in,
                              void* d_out, int out_size, void* d_ws,
                              size_t ws_size, hipStream_t stream) {
  (void)in_sizes; (void)n_in; (void)out_size; (void)ws_size;

  const float* x      = (const float*)d_in[0];
  const float* enc    = (const float*)d_in[1];
  const float* sa_wq  = (const float*)d_in[2];
  const float* sa_wk  = (const float*)d_in[3];
  const float* sa_wv  = (const float*)d_in[4];
  const float* sa_wo  = (const float*)d_in[5];
  const float* sa_bo  = (const float*)d_in[6];
  const float* ca_wq  = (const float*)d_in[7];
  const float* ca_wk  = (const float*)d_in[8];
  const float* ca_wv  = (const float*)d_in[9];
  const float* ca_wo  = (const float*)d_in[10];
  const float* ca_bo  = (const float*)d_in[11];
  const float* ln1_g  = (const float*)d_in[12];
  const float* ln1_b  = (const float*)d_in[13];
  const float* ln2_g  = (const float*)d_in[14];
  const float* ln2_b  = (const float*)d_in[15];
  const float* ln3_g  = (const float*)d_in[16];
  const float* ln3_b  = (const float*)d_in[17];
  const float* fc1_w  = (const float*)d_in[18];
  const float* fc1_b  = (const float*)d_in[19];
  const float* fc2_w  = (const float*)d_in[20];
  const float* fc2_b  = (const float*)d_in[21];
  const unsigned char* tgt_mask = (const unsigned char*)d_in[22];  // jnp bool_
  const unsigned char* src_mask = (const unsigned char*)d_in[23];

  // Deterministic bump allocator over d_ws (~176 MB total).
  char* ws = (char*)d_ws;
  size_t off = 0;
  auto alloc = [&](size_t bytes) -> char* {
    char* p = ws + off;
    off += (bytes + 255) & ~(size_t)255;
    return p;
  };

  __bf16* x_bf   = (__bf16*)alloc((size_t)M_ * D_ * 2);
  __bf16* enc_bf = (__bf16*)alloc((size_t)M_ * D_ * 2);
  __bf16* sa_wq_t = (__bf16*)alloc((size_t)D_ * INNER_ * 2);
  __bf16* sa_wk_t = (__bf16*)alloc((size_t)D_ * INNER_ * 2);
  __bf16* sa_wv_t = (__bf16*)alloc((size_t)D_ * INNER_ * 2);
  __bf16* sa_wo_t = (__bf16*)alloc((size_t)INNER_ * D_ * 2);
  __bf16* ca_wq_t = (__bf16*)alloc((size_t)D_ * INNER_ * 2);
  __bf16* ca_wk_t = (__bf16*)alloc((size_t)D_ * INNER_ * 2);
  __bf16* ca_wv_t = (__bf16*)alloc((size_t)D_ * INNER_ * 2);
  __bf16* ca_wo_t = (__bf16*)alloc((size_t)INNER_ * D_ * 2);
  __bf16* fc1_t   = (__bf16*)alloc((size_t)D_ * FF_ * 2);
  __bf16* fc2_t   = (__bf16*)alloc((size_t)FF_ * D_ * 2);
  __bf16* qb    = (__bf16*)alloc((size_t)M_ * INNER_ * 2);
  __bf16* kb    = (__bf16*)alloc((size_t)M_ * INNER_ * 2);
  __bf16* vb    = (__bf16*)alloc((size_t)M_ * INNER_ * 2);
  __bf16* attnb = (__bf16*)alloc((size_t)M_ * INNER_ * 2);
  float*  tmp   = (float*) alloc((size_t)M_ * D_ * 4);
  float*  x1f   = (float*) alloc((size_t)M_ * D_ * 4);
  float*  x2f   = (float*) alloc((size_t)M_ * D_ * 4);
  __bf16* x1b   = (__bf16*)alloc((size_t)M_ * D_ * 2);
  __bf16* x2b   = (__bf16*)alloc((size_t)M_ * D_ * 2);
  __bf16* hb    = (__bf16*)alloc((size_t)M_ * FF_ * 2);

  // ---- precision conversions ----
  to_bf16<<<1024, 256, 0, stream>>>(x,   x_bf,   (size_t)M_ * D_);
  to_bf16<<<1024, 256, 0, stream>>>(enc, enc_bf, (size_t)M_ * D_);
  transpose_to_bf16<<<dim3(INNER_/32, D_/32), 256, 0, stream>>>(sa_wq, sa_wq_t, D_, INNER_);
  transpose_to_bf16<<<dim3(INNER_/32, D_/32), 256, 0, stream>>>(sa_wk, sa_wk_t, D_, INNER_);
  transpose_to_bf16<<<dim3(INNER_/32, D_/32), 256, 0, stream>>>(sa_wv, sa_wv_t, D_, INNER_);
  transpose_to_bf16<<<dim3(D_/32, INNER_/32), 256, 0, stream>>>(sa_wo, sa_wo_t, INNER_, D_);
  transpose_to_bf16<<<dim3(INNER_/32, D_/32), 256, 0, stream>>>(ca_wq, ca_wq_t, D_, INNER_);
  transpose_to_bf16<<<dim3(INNER_/32, D_/32), 256, 0, stream>>>(ca_wk, ca_wk_t, D_, INNER_);
  transpose_to_bf16<<<dim3(INNER_/32, D_/32), 256, 0, stream>>>(ca_wv, ca_wv_t, D_, INNER_);
  transpose_to_bf16<<<dim3(D_/32, INNER_/32), 256, 0, stream>>>(ca_wo, ca_wo_t, INNER_, D_);
  transpose_to_bf16<<<dim3(FF_/32, D_/32),    256, 0, stream>>>(fc1_w, fc1_t,   D_, FF_);
  transpose_to_bf16<<<dim3(D_/32, FF_/32),    256, 0, stream>>>(fc2_w, fc2_t,   FF_, D_);

  const dim3 gProj(INNER_/128, M_/128);   // (8, 32)
  const dim3 gD   (D_/128,     M_/128);
  const dim3 gFF  (FF_/128,    M_/128);
  const dim3 gAttn(S_/64, H_, B_);

  // ---- self attention ----
  gemm_wmma<0, __bf16><<<gProj, 256, 0, stream>>>(x_bf, sa_wq_t, nullptr, qb, M_, INNER_, D_);
  gemm_wmma<0, __bf16><<<gProj, 256, 0, stream>>>(x_bf, sa_wk_t, nullptr, kb, M_, INNER_, D_);
  gemm_wmma<0, __bf16><<<gProj, 256, 0, stream>>>(x_bf, sa_wv_t, nullptr, vb, M_, INNER_, D_);
  attn_wmma<<<gAttn, 128, 0, stream>>>(qb, kb, vb, tgt_mask, attnb);
  gemm_wmma<1, float><<<gD, 256, 0, stream>>>(attnb, sa_wo_t, sa_bo, tmp, M_, D_, INNER_);
  add_ln<<<M_, 256, 0, stream>>>(x, tmp, ln1_g, ln1_b, x1f, x1b);

  // ---- cross attention ----
  gemm_wmma<0, __bf16><<<gProj, 256, 0, stream>>>(x1b,    ca_wq_t, nullptr, qb, M_, INNER_, D_);
  gemm_wmma<0, __bf16><<<gProj, 256, 0, stream>>>(enc_bf, ca_wk_t, nullptr, kb, M_, INNER_, D_);
  gemm_wmma<0, __bf16><<<gProj, 256, 0, stream>>>(enc_bf, ca_wv_t, nullptr, vb, M_, INNER_, D_);
  attn_wmma<<<gAttn, 128, 0, stream>>>(qb, kb, vb, src_mask, attnb);
  gemm_wmma<1, float><<<gD, 256, 0, stream>>>(attnb, ca_wo_t, ca_bo, tmp, M_, D_, INNER_);
  add_ln<<<M_, 256, 0, stream>>>(x1f, tmp, ln2_g, ln2_b, x2f, x2b);

  // ---- MLP ----
  gemm_wmma<2, __bf16><<<gFF, 256, 0, stream>>>(x2b, fc1_t, fc1_b, hb,  M_, FF_, D_);
  gemm_wmma<1, float><<<gD,  256, 0, stream>>>(hb,  fc2_t, fc2_b, tmp, M_, D_, FF_);
  add_ln<<<M_, 256, 0, stream>>>(x2f, tmp, ln3_g, ln3_b, (float*)d_out, nullptr);
}
